// Net_13142599925794
// MI455X (gfx1250) — matrix operand order repaired and checked by reading, not compiled, not run on hardware
//
#include <hip/hip_runtime.h>

// Problem constants (from reference)
#define HH    64      // hidden size
#define G4    256     // 4*H gate width
#define TLEN  512     // sequence length
#define NB    1024    // batch
#define NTILE 16      // batch rows per workgroup
#define XCHUNK 64     // timesteps of x staged in LDS per refill

typedef __attribute__((ext_vector_type(16))) _Float16 v16h;
typedef __attribute__((ext_vector_type(8)))  float    v8f;
typedef __attribute__((ext_vector_type(8)))  _Float16 h8;

union V16u { v16h v; h8 h[2]; };

// ---- WMMA operand builders (CDNA5 16x16x32 f16 layouts, wave32) ----
// A: 16x32, lane&15 = row M, lane>>4 selects K half-group (+8).
// halves [k0..k0+7] at +0 and [k0+16..k0+23] at +16, k0 = kofs + (lane>>4)*8.
__device__ __forceinline__ v16h load_a(const _Float16* base, int stride, int kofs, int lane) {
  const int m  = lane & 15;
  const int hi = lane >> 4;
  const _Float16* p = base + m * stride + kofs + hi * 8;
  V16u u;
  u.h[0] = *(const h8*)(p);
  u.h[1] = *(const h8*)(p + 16);
  return u.v;
}

// B: 32x16, lane&15 = col N, lane>>4 selects K half-group (+8).
// halves [k0..k0+7] at +0 and [k0+16..k0+23] at +16, k0 = kofs + (lane>>4)*8.  W row-major [n][K].
__device__ __forceinline__ v16h load_b(const _Float16* W, int K, int colbase, int kofs, int lane) {
  const int n  = colbase + (lane & 15);
  const int hi = lane >> 4;
  const _Float16* p = W + n * K + kofs + hi * 8;
  V16u u;
  u.h[0] = *(const h8*)(p);
  u.h[1] = *(const h8*)(p + 16);
  return u.v;
}

__device__ __forceinline__ float fsig(float x) {
  return __builtin_amdgcn_rcpf(1.0f + __expf(-x));
}
__device__ __forceinline__ float ftanh(float x) {
  float e = __expf(-2.0f * x);
  return (1.0f - e) * __builtin_amdgcn_rcpf(1.0f + e);
}

__device__ __forceinline__ void st4(float* p, float a, float b, float c, float d) {
  *(float4*)p = make_float4(a, b, c, d);
}

#define WMMA(A, B, C) __builtin_amdgcn_wmma_f32_16x16x32_f16(false, (A), false, (B), (short)0, (C), false, false)

// ---- dynamic LDS layout ----
constexpr size_t SMEM_HALVES = 256*64 + 256*128 + 256*128 + 32*64   // weight staging
                             + 16*128 + 16*128 + 16*32;             // hcat2, hcat3, y1
constexpr size_t SMEM_FLOATS = 256*16            // gcm: gates column-major [n][m]
                             + 3*(16*64)         // c1,c2,c3  ([u][m] == flat e)
                             + 4*256             // b1s,b2s,b3s,wih1s
                             + 32 + 32 + 1       // fc1bs, fc2ws, fc2bs
                             + XCHUNK*16;        // xbufT [t_local][m]
constexpr size_t SMEM_BYTES  = SMEM_HALVES * 2 + SMEM_FLOATS * 4;

extern "C" __global__ __launch_bounds__(256)
void lstm3_persistent(const float* __restrict__ x,
                      const float* __restrict__ w_ih1, const float* __restrict__ w_hh1, const float* __restrict__ b1,
                      const float* __restrict__ w_ih2, const float* __restrict__ w_hh2, const float* __restrict__ b2,
                      const float* __restrict__ w_ih3, const float* __restrict__ w_hh3, const float* __restrict__ b3,
                      const float* __restrict__ fc1_w, const float* __restrict__ fc1_b,
                      const float* __restrict__ fc2_w, const float* __restrict__ fc2_b,
                      float* __restrict__ out)
{
  extern __shared__ __align__(16) char smem_raw[];
  _Float16* Wg1   = (_Float16*)smem_raw;        // 256*64
  _Float16* Wg2   = Wg1   + 256*64;             // 256*128
  _Float16* Wg3   = Wg2   + 256*128;            // 256*128
  _Float16* Wfc1  = Wg3   + 256*128;            // 32*64
  _Float16* hcat2 = Wfc1  + 32*64;              // 16*128  [h1 | h2]
  _Float16* hcat3 = hcat2 + 16*128;             // 16*128  [h2 | h3]
  _Float16* y1    = hcat3 + 16*128;             // 16*32
  float* gcm   = (float*)(y1 + 16*32);          // 256*16 gates, column-major [n][m]
  float* c1    = gcm   + 256*16;                // 16*64 (flat e = u*16+m)
  float* c2    = c1    + 16*64;
  float* c3    = c2    + 16*64;
  float* b1s   = c3    + 16*64;                 // 256
  float* b2s   = b1s   + 256;
  float* b3s   = b2s   + 256;
  float* wih1s = b3s   + 256;                   // 256
  float* fc1bs = wih1s + 256;                   // 32
  float* fc2ws = fc1bs + 32;                    // 32
  float* fc2bs = fc2ws + 32;                    // 1
  float* xbufT = fc2bs + 1;                     // XCHUNK*16  [t_local][m]

  const int tid  = threadIdx.x;
  const int wave = tid >> 5;
  const int lane = tid & 31;
  const int m_of = (lane >> 4) * 8;   // D-row offset for this lane half
  const int nn   = lane & 15;         // D-column within tile
  const int rowbase = blockIdx.x * NTILE;

  // ---- one-time weight pack: f32 global -> f16 LDS (row-major [out][K]) ----
  for (int i = tid; i < 256*64; i += 256) Wg1[i] = (_Float16)w_hh1[i];
  for (int i = tid; i < 256*128; i += 256) {
    int j = i >> 7, k = i & 127;
    Wg2[i] = (_Float16)((k < 64) ? w_ih2[j*64 + k] : w_hh2[j*64 + (k-64)]);
    Wg3[i] = (_Float16)((k < 64) ? w_ih3[j*64 + k] : w_hh3[j*64 + (k-64)]);
  }
  for (int i = tid; i < 32*64; i += 256) Wfc1[i] = (_Float16)fc1_w[i];
  if (tid < 256) { b1s[tid] = b1[tid]; b2s[tid] = b2[tid]; b3s[tid] = b3[tid]; wih1s[tid] = w_ih1[tid]; }
  if (tid < 32)  { fc1bs[tid] = fc1_b[tid]; fc2ws[tid] = fc2_w[tid]; }
  if (tid == 0)  fc2bs[0] = fc2_b[0];
  for (int i = tid; i < 16*128; i += 256) { hcat2[i] = (_Float16)0.f; hcat3[i] = (_Float16)0.f; }
  for (int i = tid; i < 16*64;  i += 256) { c1[i] = 0.f; c2[i] = 0.f; c3[i] = 0.f; }
  __syncthreads();

  // ---- hoist weights into registers (loop-invariant over T) ----
  v16h B1r[2][2], B2r[4][2], B3r[4][2], Bfr[2];
  #pragma unroll
  for (int ks = 0; ks < 2; ++ks)
    #pragma unroll
    for (int tl = 0; tl < 2; ++tl)
      B1r[ks][tl] = load_b(Wg1, 64, wave*32 + tl*16, ks*32, lane);
  #pragma unroll
  for (int ks = 0; ks < 4; ++ks)
    #pragma unroll
    for (int tl = 0; tl < 2; ++tl) {
      B2r[ks][tl] = load_b(Wg2, 128, wave*32 + tl*16, ks*32, lane);
      B3r[ks][tl] = load_b(Wg3, 128, wave*32 + tl*16, ks*32, lane);
    }
  if (wave < 2) {
    #pragma unroll
    for (int ks = 0; ks < 2; ++ks)
      Bfr[ks] = load_b(Wfc1, 64, wave*16, ks*32, lane);
  }
  // per-lane loop-invariant scalars
  const int n0 = wave*32 + nn, n1 = n0 + 16;
  const float bn1_0 = b1s[n0], bn1_1 = b1s[n1];
  const float wn1_0 = wih1s[n0], wn1_1 = wih1s[n1];
  const float bn2_0 = b2s[n0], bn2_1 = b2s[n1];
  const float bn3_0 = b3s[n0], bn3_1 = b3s[n1];
  const float bfc   = (wave < 2) ? fc1bs[wave*16 + nn] : 0.f;
  const float fc2b_r = fc2bs[0];

  for (int t = 0; t < TLEN; ++t) {
    // ---- refill x chunk (transposed [t_local][m]) every XCHUNK steps ----
    if ((t & (XCHUNK - 1)) == 0) {
      const int i0 = tid * 4;            // 0..1020
      const int m = i0 >> 6, col = i0 & 63;
      float4 v = *(const float4*)&x[(size_t)(rowbase + m) * TLEN + t + col];
      xbufT[(col+0)*16 + m] = v.x;
      xbufT[(col+1)*16 + m] = v.y;
      xbufT[(col+2)*16 + m] = v.z;
      xbufT[(col+3)*16 + m] = v.w;
      __syncthreads();
    }

    // ---- Layer 1: gates = h1 @ w_hh1^T + x_t * w_ih1 + b1   (K=64) ----
    {
      v8f a0 = {}, a1 = {};
      #pragma unroll
      for (int ks = 0; ks < 2; ++ks) {
        v16h a = load_a(hcat2, 128, ks * 32, lane);     // h1 region
        a0 = WMMA(a, B1r[ks][0], a0);
        a1 = WMMA(a, B1r[ks][1], a1);
      }
      const float* xp = &xbufT[(t & (XCHUNK - 1)) * 16 + m_of];
      float4 xlo = *(const float4*)xp;
      float4 xhi = *(const float4*)(xp + 4);
      st4(&gcm[n0*16 + m_of],     a0[0]+bn1_0+xlo.x*wn1_0, a0[1]+bn1_0+xlo.y*wn1_0,
                                  a0[2]+bn1_0+xlo.z*wn1_0, a0[3]+bn1_0+xlo.w*wn1_0);
      st4(&gcm[n0*16 + m_of + 4], a0[4]+bn1_0+xhi.x*wn1_0, a0[5]+bn1_0+xhi.y*wn1_0,
                                  a0[6]+bn1_0+xhi.z*wn1_0, a0[7]+bn1_0+xhi.w*wn1_0);
      st4(&gcm[n1*16 + m_of],     a1[0]+bn1_1+xlo.x*wn1_1, a1[1]+bn1_1+xlo.y*wn1_1,
                                  a1[2]+bn1_1+xlo.z*wn1_1, a1[3]+bn1_1+xlo.w*wn1_1);
      st4(&gcm[n1*16 + m_of + 4], a1[4]+bn1_1+xhi.x*wn1_1, a1[5]+bn1_1+xhi.y*wn1_1,
                                  a1[6]+bn1_1+xhi.z*wn1_1, a1[7]+bn1_1+xhi.w*wn1_1);
    }
    __syncthreads();
    // elementwise LSTM update, layer 1  (e = u*16 + m)
    #pragma unroll
    for (int q = 0; q < 4; ++q) {
      const int e = tid + q*256, u = e >> 4, m = e & 15;
      const float gi = fsig (gcm[         e]);
      const float gf = fsig (gcm[ 64*16 + e]);
      const float gg = ftanh(gcm[128*16 + e]);
      const float go = fsig (gcm[192*16 + e]);
      const float c = gf * c1[e] + gi * gg;
      c1[e] = c;
      hcat2[m*128 + u] = (_Float16)(go * ftanh(c));   // h1
    }
    __syncthreads();

    // ---- Layer 2: gates = [h1 | h2] @ [w_ih2 | w_hh2]^T + b2   (K=128) ----
    {
      v8f a0 = {}, a1 = {};
      #pragma unroll
      for (int ks = 0; ks < 4; ++ks) {
        v16h a = load_a(hcat2, 128, ks * 32, lane);
        a0 = WMMA(a, B2r[ks][0], a0);
        a1 = WMMA(a, B2r[ks][1], a1);
      }
      st4(&gcm[n0*16 + m_of],     a0[0]+bn2_0, a0[1]+bn2_0, a0[2]+bn2_0, a0[3]+bn2_0);
      st4(&gcm[n0*16 + m_of + 4], a0[4]+bn2_0, a0[5]+bn2_0, a0[6]+bn2_0, a0[7]+bn2_0);
      st4(&gcm[n1*16 + m_of],     a1[0]+bn2_1, a1[1]+bn2_1, a1[2]+bn2_1, a1[3]+bn2_1);
      st4(&gcm[n1*16 + m_of + 4], a1[4]+bn2_1, a1[5]+bn2_1, a1[6]+bn2_1, a1[7]+bn2_1);
    }
    __syncthreads();
    #pragma unroll
    for (int q = 0; q < 4; ++q) {
      const int e = tid + q*256, u = e >> 4, m = e & 15;
      const float gi = fsig (gcm[         e]);
      const float gf = fsig (gcm[ 64*16 + e]);
      const float gg = ftanh(gcm[128*16 + e]);
      const float go = fsig (gcm[192*16 + e]);
      const float c = gf * c2[e] + gi * gg;
      c2[e] = c;
      const _Float16 h = (_Float16)(go * ftanh(c));
      hcat2[m*128 + 64 + u] = h;                      // h2 -> L2(t+1)
      hcat3[m*128 +      u] = h;                      // h2 -> L3 input
    }
    __syncthreads();

    // ---- Layer 3: gates = [h2 | h3] @ [w_ih3 | w_hh3]^T + b3   (K=128) ----
    {
      v8f a0 = {}, a1 = {};
      #pragma unroll
      for (int ks = 0; ks < 4; ++ks) {
        v16h a = load_a(hcat3, 128, ks * 32, lane);
        a0 = WMMA(a, B3r[ks][0], a0);
        a1 = WMMA(a, B3r[ks][1], a1);
      }
      st4(&gcm[n0*16 + m_of],     a0[0]+bn3_0, a0[1]+bn3_0, a0[2]+bn3_0, a0[3]+bn3_0);
      st4(&gcm[n0*16 + m_of + 4], a0[4]+bn3_0, a0[5]+bn3_0, a0[6]+bn3_0, a0[7]+bn3_0);
      st4(&gcm[n1*16 + m_of],     a1[0]+bn3_1, a1[1]+bn3_1, a1[2]+bn3_1, a1[3]+bn3_1);
      st4(&gcm[n1*16 + m_of + 4], a1[4]+bn3_1, a1[5]+bn3_1, a1[6]+bn3_1, a1[7]+bn3_1);
    }
    __syncthreads();
    #pragma unroll
    for (int q = 0; q < 4; ++q) {
      const int e = tid + q*256, u = e >> 4, m = e & 15;
      const float gi = fsig (gcm[         e]);
      const float gf = fsig (gcm[ 64*16 + e]);
      const float gg = ftanh(gcm[128*16 + e]);
      const float go = fsig (gcm[192*16 + e]);
      const float c = gf * c3[e] + gi * gg;
      c3[e] = c;
      hcat3[m*128 + 64 + u] = (_Float16)(go * ftanh(c));  // h3
    }
    __syncthreads();

    // ---- fc1: y1 = leaky_relu(h3 @ fc1_w^T + fc1_b)  [16x32], waves 0..1 ----
    if (wave < 2) {
      v8f acc = {};
      #pragma unroll
      for (int ks = 0; ks < 2; ++ks) {
        v16h a = load_a(hcat3 + 64, 128, ks * 32, lane);  // h3 region
        acc = WMMA(a, Bfr[ks], acc);
      }
      const int n = wave*16 + nn;
      #pragma unroll
      for (int r = 0; r < 8; ++r) {
        float v = acc[r] + bfc;
        v = (v > 0.f) ? v : 0.2f * v;
        y1[(r + m_of)*32 + n] = (_Float16)v;
      }
    }
    __syncthreads();

    // ---- fc2: scalar head per row (32-wide dot) ----
    if (tid < 16) {
      float s = fc2b_r;
      #pragma unroll
      for (int j = 0; j < 32; ++j) s += (float)y1[tid*32 + j] * fc2ws[j];
      out[(size_t)(rowbase + tid) * TLEN + t] = s;
    }
    // no trailing barrier needed: gates/y1 writers at t+1 sit behind later syncs
  }
}

extern "C" void kernel_launch(void* const* d_in, const int* in_sizes, int n_in,
                              void* d_out, int out_size, void* d_ws, size_t ws_size,
                              hipStream_t stream) {
  (void)in_sizes; (void)n_in; (void)out_size; (void)d_ws; (void)ws_size;
  const float* x     = (const float*)d_in[0];
  const float* w_ih1 = (const float*)d_in[1];
  const float* w_hh1 = (const float*)d_in[2];
  const float* b1    = (const float*)d_in[3];
  const float* w_ih2 = (const float*)d_in[4];
  const float* w_hh2 = (const float*)d_in[5];
  const float* b2    = (const float*)d_in[6];
  const float* w_ih3 = (const float*)d_in[7];
  const float* w_hh3 = (const float*)d_in[8];
  const float* b3    = (const float*)d_in[9];
  const float* fc1_w = (const float*)d_in[10];
  const float* fc1_b = (const float*)d_in[11];
  const float* fc2_w = (const float*)d_in[12];
  const float* fc2_b = (const float*)d_in[13];
  float* out = (float*)d_out;

  (void)hipFuncSetAttribute((const void*)lstm3_persistent,
                            hipFuncAttributeMaxDynamicSharedMemorySize,
                            (int)SMEM_BYTES);

  lstm3_persistent<<<dim3(NB / NTILE), dim3(256), SMEM_BYTES, stream>>>(
      x, w_ih1, w_hh1, b1, w_ih2, w_hh2, b2, w_ih3, w_hh3, b3,
      fc1_w, fc1_b, fc2_w, fc2_b, out);
}
